// DACA_81140522156635
// MI455X (gfx1250) — matrix operand order repaired and checked by reading, not compiled, not run on hardware
//
#include <hip/hip_runtime.h>

// Problem constants (from reference)
#define Bb   8
#define Cc   256
#define Hh   64
#define Ww   64
#define Mm   4
#define Rr   4
#define TEMB 128
#define HW   (Hh*Ww)      // 4096
#define NPIX (Bb*HW)      // 32768 GEMM-M
#define K2C  (2*Cc)       // 512 conv in-channels
#define KF   (Mm*Cc)      // 1024 GEMM-K
#define NOFF (Bb*2*Mm*HW) // 262144 offset elements

typedef __attribute__((ext_vector_type(16))) __bf16       v16bf;
typedef __attribute__((ext_vector_type(8)))  float        v8f;
typedef __attribute__((ext_vector_type(4)))  float        f32x4;
typedef __attribute__((ext_vector_type(4)))  unsigned int u32x4;

union AV { v16bf v; u32x4 q[2]; };

__device__ __forceinline__ unsigned short f2bf(float f) {
    unsigned int u = __float_as_uint(f);
    unsigned int r = u + 0x7FFFu + ((u >> 16) & 1u);   // round-to-nearest-even
    return (unsigned short)(r >> 16);
}

__device__ __forceinline__ v8f wmma_bf16(const AV& a, const AV& b, v8f c) {
    return __builtin_amdgcn_wmma_f32_16x16x32_bf16(
        false, a.v, false, b.v, (short)0, c, false, false);
}

// CDNA5 async global->LDS copy (16B per lane) + wait, via inline asm
// (portable across both toolchains per bridge doc; builtin arity differs).
__device__ __forceinline__ void async_copy_b128(unsigned lds_byte_addr, const void* gaddr) {
    asm volatile("global_load_async_to_lds_b128 %0, %1, off"
                 :: "v"(lds_byte_addr), "v"(gaddr) : "memory");
}
__device__ __forceinline__ void wait_asynccnt0() {
    asm volatile("s_wait_asynccnt 0x0" ::: "memory");
}

__global__ void zero_kernel(float* p) { p[0] = 0.f; }

// ---------------------------------------------------------------------------
// 12-channel 3x3 conv over concat(F_x,F_c): offsets (8ch) + sigmoid attn (4ch)
// One block per (b,h) row; LDS stages 3 input rows per channel.
// Also accumulates sum(|offsets|) for the loss.
// ---------------------------------------------------------------------------
__global__ void conv12_kernel(const float* __restrict__ Fx, const float* __restrict__ Fc,
                              const float* __restrict__ w_off, const float* __restrict__ b_off,
                              const float* __restrict__ w_attn, const float* __restrict__ b_attn,
                              float* __restrict__ out_off, float* __restrict__ attn_ws,
                              float* __restrict__ loss_acc) {
    __shared__ float rows[3][66];   // 3 input rows, halo col 0 <-> x=-1
    __shared__ float wl[108];       // 12 outc x 9 taps for current channel
    __shared__ float red[256];
    const int t  = threadIdx.x;
    const int b  = blockIdx.x / Hh;
    const int h  = blockIdx.x % Hh;
    const int w  = t & 63;
    const int og = t >> 6;          // output-channel group 0..3 -> outc og*3..og*3+2
    float acc[3] = {0.f, 0.f, 0.f};

    for (int c = 0; c < K2C; ++c) {
        __syncthreads();
        const float* src = (c < Cc) ? (Fx + ((size_t)(b*Cc + c))*HW)
                                    : (Fc + ((size_t)(b*Cc + (c - Cc)))*HW);
        if (t < 198) {
            int ry = t / 66, rx = t % 66;
            int y = h - 1 + ry, x = rx - 1;
            float v = 0.f;
            if (y >= 0 && y < Hh && x >= 0 && x < Ww) v = src[y*Ww + x];
            rows[ry][rx] = v;
        }
        if (t < 108) {
            int oc = t / 9, tap = t % 9;
            wl[t] = (oc < 8) ? w_off[(size_t)(oc*K2C + c)*9 + tap]
                             : w_attn[(size_t)((oc - 8)*K2C + c)*9 + tap];
        }
        __syncthreads();
        #pragma unroll
        for (int k = 0; k < 3; ++k) {
            const int oc = og*3 + k;
            const float* wp = &wl[oc*9];
            float s = 0.f;
            #pragma unroll
            for (int ky = 0; ky < 3; ++ky)
                #pragma unroll
                for (int kx = 0; kx < 3; ++kx)
                    s += wp[ky*3 + kx] * rows[ky][w + kx];
            acc[k] += s;
        }
    }

    float labs = 0.f;
    const int hw = h*Ww + w;
    #pragma unroll
    for (int k = 0; k < 3; ++k) {
        const int oc = og*3 + k;
        float v = acc[k] + ((oc < 8) ? b_off[oc] : b_attn[oc - 8]);
        if (oc < 8) {   // offsets layout [B, M, 2, H, W] -> channel oc = m*2+comp
            out_off[(size_t)(b*8 + oc)*HW + hw] = v;
            labs += fabsf(v);
        } else {
            attn_ws[(size_t)(b*Mm + (oc - 8))*HW + hw] = 1.f/(1.f + expf(-v));
        }
    }
    red[t] = labs;
    __syncthreads();
    for (int s = 128; s > 0; s >>= 1) { if (t < s) red[t] += red[t + s]; __syncthreads(); }
    if (t == 0) atomicAdd(loss_acc, red[0]);
}

// ---------------------------------------------------------------------------
// alpha = sigmoid(relu(t*w1 + b1) @ w2 + b2), per batch
// ---------------------------------------------------------------------------
__global__ void alpha_kernel(const float* __restrict__ t_in, const float* __restrict__ w1,
                             const float* __restrict__ b1, const float* __restrict__ w2,
                             const float* __restrict__ b2, float* __restrict__ alpha_ws) {
    __shared__ float red[TEMB];
    const int b = blockIdx.x, j = threadIdx.x;
    float hv = fmaxf(t_in[b]*w1[j] + b1[j], 0.f);
    red[j] = hv * w2[j];
    __syncthreads();
    for (int s = 64; s > 0; s >>= 1) { if (j < s) red[j] += red[j + s]; __syncthreads(); }
    if (j == 0) alpha_ws[b] = 1.f/(1.f + expf(-(red[0] + b2[0])));
}

// ---------------------------------------------------------------------------
// Repack w_fuse [256 x 1024] into bf16 WMMA-B tiles:
//   bpack[((nt*32 + kg)*32 + lane)*16 + j] = w_fuse[n][k]
//   n = nt*16 + (lane&15),  k = kg*32 + (lane>>4)*16 + j
// ---------------------------------------------------------------------------
__global__ void prep_b_kernel(const float* __restrict__ w_fuse, unsigned short* __restrict__ bpack) {
    const int f    = blockIdx.x*blockDim.x + threadIdx.x;  // < 262144
    const int j    = f & 15;
    const int lane = (f >> 4) & 31;
    const int kg   = (f >> 9) & 31;
    const int nt   = f >> 14;
    const int k = kg*32 + (lane >> 4)*16 + j;
    const int n = nt*16 + (lane & 15);
    bpack[f] = f2bf(w_fuse[(size_t)n*KF + k]);
}

// ---------------------------------------------------------------------------
// Transpose F_c NCHW -> [b][hw][c] so bilinear gathers become channel-vector
// loads. 32x32 LDS tile, coalesced both sides.
// ---------------------------------------------------------------------------
__global__ void transpose_fc_kernel(const float* __restrict__ Fc, float* __restrict__ Fct) {
    __shared__ float tile[32][33];
    const int b   = blockIdx.z;
    const int c0  = blockIdx.x*32;
    const int hw0 = blockIdx.y*32;
    const int tx = threadIdx.x;   // 32
    const int ty = threadIdx.y;   // 8
    for (int i = ty; i < 32; i += 8)
        tile[i][tx] = Fc[((size_t)(b*Cc + c0 + i))*HW + hw0 + tx];
    __syncthreads();
    for (int i = ty; i < 32; i += 8)
        Fct[((size_t)(b*HW + hw0 + i))*Cc + c0 + tx] = tile[tx][i];
}

// ---------------------------------------------------------------------------
// Sampler: head[pixel][m*256+c] = attn * sum_r Wp[m,r] * bilinear(F_ct, grid)
// Faithful to reference: grid comp0 (ys[h]+bp_y+off0) consumed as gx,
// comp1 (xs[w]+bp_x+off1) as gy (preserved x/y swap). Channel-contiguous
// Fct makes each corner gather a float4 of 4 channels; head stored as
// packed 4x bf16 b64 writes.
// ---------------------------------------------------------------------------
__global__ void sampler_kernel(const float* __restrict__ Fct, const float* __restrict__ offs,
                               const float* __restrict__ attn_ws, const float* __restrict__ Wp,
                               unsigned short* __restrict__ head) {
    const int bid = blockIdx.x;
    const int b = bid / (Mm*Hh);
    const int m = (bid / Hh) % Mm;
    const int h = bid % Hh;
    const int w = threadIdx.x;       // 0..63
    const int part = threadIdx.y;    // 0..3
    const int hw = h*Ww + w;

    const float off0 = offs[(size_t)(b*8 + m*2 + 0)*HW + hw];
    const float off1 = offs[(size_t)(b*8 + m*2 + 1)*HW + hw];
    const float av   = attn_ws[(size_t)(b*Mm + m)*HW + hw];
    const float ysh  = -1.f + 2.f*(float)h/63.f;
    const float xsw  = -1.f + 2.f*(float)w/63.f;
    const float bpy[4] = {-0.5f, -0.5f, 0.5f, 0.5f};
    const float bpx[4] = {-0.5f,  0.5f, -0.5f, 0.5f};

    const float* cp[16];
    float wgt[16];
    const float* fb = Fct + ((size_t)b*HW)*Cc + part*64;
    #pragma unroll
    for (int r = 0; r < Rr; ++r) {
        const float wr = Wp[m*Rr + r];
        const float g0 = ysh + bpy[r] + off0;      // consumed as x (faithful)
        const float g1 = xsw + bpx[r] + off1;      // consumed as y
        const float gx = (g0 + 1.f)*0.5f*(float)(Ww - 1);
        const float gy = (g1 + 1.f)*0.5f*(float)(Hh - 1);
        const float x0f = floorf(gx), y0f = floorf(gy);
        const float wx1 = gx - x0f, wx0 = 1.f - wx1;
        const float wy1 = gy - y0f, wy0 = 1.f - wy1;
        const int x0 = (int)x0f, y0 = (int)y0f;
        const int x1 = x0 + 1,  y1 = y0 + 1;
        const bool vx0 = (x0 >= 0) && (x0 < Ww), vx1 = (x1 >= 0) && (x1 < Ww);
        const bool vy0 = (y0 >= 0) && (y0 < Hh), vy1 = (y1 >= 0) && (y1 < Hh);
        const int xc0 = min(max(x0, 0), Ww - 1), xc1 = min(max(x1, 0), Ww - 1);
        const int yc0 = min(max(y0, 0), Hh - 1), yc1 = min(max(y1, 0), Hh - 1);
        cp[r*4+0] = fb + (size_t)(yc0*Ww + xc0)*Cc; wgt[r*4+0] = (vy0 && vx0) ? wr*wy0*wx0 : 0.f;
        cp[r*4+1] = fb + (size_t)(yc0*Ww + xc1)*Cc; wgt[r*4+1] = (vy0 && vx1) ? wr*wy0*wx1 : 0.f;
        cp[r*4+2] = fb + (size_t)(yc1*Ww + xc0)*Cc; wgt[r*4+2] = (vy1 && vx0) ? wr*wy1*wx0 : 0.f;
        cp[r*4+3] = fb + (size_t)(yc1*Ww + xc1)*Cc; wgt[r*4+3] = (vy1 && vx1) ? wr*wy1*wx1 : 0.f;
    }

    unsigned short* hdst = head + ((size_t)(b*HW + hw))*KF + m*Cc + part*64;
    for (int cq = 0; cq < 16; ++cq) {          // 4 channels per step
        f32x4 s = {0.f, 0.f, 0.f, 0.f};
        #pragma unroll
        for (int i = 0; i < 16; ++i) {
            const f32x4 v = *(const f32x4*)(cp[i] + cq*4);
            s += v * wgt[i];
        }
        s *= av;
        unsigned long long pk =  (unsigned long long)f2bf(s.x)
                              | ((unsigned long long)f2bf(s.y) << 16)
                              | ((unsigned long long)f2bf(s.z) << 32)
                              | ((unsigned long long)f2bf(s.w) << 48);
        *(unsigned long long*)(hdst + cq*4) = pk;
    }
}

// ---------------------------------------------------------------------------
// Fuse GEMM via v_wmma_f32_16x16x32_bf16.
// Block: 8 waves = 256 pixels x 64 out-channels. B panel (128KB) staged
// once into LDS with CDNA5 async global->LDS copies; WMMA B operands then
// come from ds_load_b128. Wave tile: 32 pixels x 64 co (8 accumulators).
// Epilogue fuses bias + alpha blend with F_x.
// ---------------------------------------------------------------------------
__global__ __launch_bounds__(256) void fuse_gemm_kernel(
        const unsigned short* __restrict__ head,
        const unsigned short* __restrict__ bpack,
        const float* __restrict__ Fx,
        const float* __restrict__ b_fuse,
        const float* __restrict__ alpha_ws,
        float* __restrict__ out) {
    __shared__ unsigned short smB[4*32*32*16];   // 128 KB B panel
    const int t    = threadIdx.x;
    const int lane = t & 31;
    const int wave = t >> 5;
    const int pix_base = blockIdx.x*256 + wave*32;
    const int co_base  = blockIdx.y*64;
    const int nt0 = co_base >> 4;
    const int hi  = lane >> 4;
    const int lo  = lane & 15;

    // Async-stage the B panel: 131072 B = 256 threads x 32 x 16B
    {
        const unsigned short* gB = bpack + (size_t)nt0*16384;
        #pragma unroll
        for (int i = 0; i < 32; ++i) {
            const int e = (i*256 + t)*8;   // ushort index, 16B granules
            async_copy_b128((unsigned)(uintptr_t)&smB[e], gB + e);
        }
        wait_asynccnt0();
        __syncthreads();
    }

    v8f acc[2][4] = {};
    const unsigned short* arow0 = head + (size_t)(pix_base + lo)*KF;
    const unsigned short* arow1 = arow0 + (size_t)16*KF;

    for (int kg = 0; kg < 32; ++kg) {
        const int k0 = kg*32;
        AV a0, a1;
        a0.q[0] = *(const u32x4*)(arow0 + k0 + hi*16);
        a0.q[1] = *(const u32x4*)(arow0 + k0 + hi*16 + 8);
        a1.q[0] = *(const u32x4*)(arow1 + k0 + hi*16);
        a1.q[1] = *(const u32x4*)(arow1 + k0 + hi*16 + 8);
        __builtin_prefetch(arow0 + k0 + 32, 0, 0);   // global_prefetch_b8
        __builtin_prefetch(arow1 + k0 + 32, 0, 0);
        #pragma unroll
        for (int j = 0; j < 4; ++j) {
            const unsigned short* bp = &smB[((j*32 + kg)*32 + lane)*16];
            AV bm;
            bm.q[0] = *(const u32x4*)(bp);
            bm.q[1] = *(const u32x4*)(bp + 8);
            acc[0][j] = wmma_bf16(a0, bm, acc[0][j]);
            acc[1][j] = wmma_bf16(a1, bm, acc[1][j]);
        }
    }

    // Epilogue: D element i of vgpr group <-> pixel row (i + hi*8), col = lo.
    const int bb  = pix_base / HW;       // uniform per block (256 | 4096)
    const int hwb = pix_base % HW;
    const float al = alpha_ws[bb];
    const float om = 1.f - al;
    #pragma unroll
    for (int p = 0; p < 2; ++p) {
        const int hwp = hwb + p*16 + hi*8;
        #pragma unroll
        for (int j = 0; j < 4; ++j) {
            const int col = co_base + j*16 + lo;
            const float bias = b_fuse[col];
            const size_t base = ((size_t)(bb*Cc + col))*HW + (size_t)hwp;
            #pragma unroll
            for (int i = 0; i < 8; ++i) {
                const float fe = acc[p][j][i] + bias;
                const float fx = Fx[base + i];
                out[base + i] = al*fe + om*fx;
            }
        }
    }
}

__global__ void loss_fin_kernel(const float* __restrict__ acc, float* __restrict__ out_loss) {
    out_loss[0] = acc[0] * (1.0f/(float)NOFF);
}

// ---------------------------------------------------------------------------
extern "C" void kernel_launch(void* const* d_in, const int* in_sizes, int n_in,
                              void* d_out, int out_size, void* d_ws, size_t ws_size,
                              hipStream_t stream) {
    (void)in_sizes; (void)n_in; (void)out_size; (void)ws_size;
    const float* Fx     = (const float*)d_in[0];
    const float* Fc     = (const float*)d_in[1];
    const float* t      = (const float*)d_in[2];
    const float* w_off  = (const float*)d_in[3];
    const float* b_off  = (const float*)d_in[4];
    const float* w_attn = (const float*)d_in[5];
    const float* b_attn = (const float*)d_in[6];
    const float* Wp     = (const float*)d_in[7];
    const float* w_fuse = (const float*)d_in[8];
    const float* b_fuse = (const float*)d_in[9];
    const float* w1     = (const float*)d_in[10];
    const float* b1     = (const float*)d_in[11];
    const float* w2     = (const float*)d_in[12];
    const float* b2     = (const float*)d_in[13];

    float* out      = (float*)d_out;                 // F_out [B,C,H,W]
    float* out_loss = out + (size_t)Bb*Cc*HW;        // scalar
    float* out_offs = out_loss + 1;                  // offsets [B,M,2,H,W]

    // Workspace layout (bytes)
    char* ws = (char*)d_ws;
    unsigned short* head     = (unsigned short*)ws;                        // 64 MB bf16 [NPIX][KF]
    unsigned short* bpack    = (unsigned short*)(ws + 67108864);           // 512 KB bf16 B tiles
    float*          attn_ws  = (float*)(ws + 67633152);                    // 512 KB [B,M,H,W]
    float*          alpha_ws = (float*)(ws + 68157440);                    // 8 floats
    float*          loss_acc = alpha_ws + 8;                               // 1 float
    float*          Fct      = (float*)(ws + 68157568);                    // 32 MB [B][HW][C]

    zero_kernel<<<1, 1, 0, stream>>>(loss_acc);
    conv12_kernel<<<dim3(Bb*Hh), dim3(256), 0, stream>>>(
        Fx, Fc, w_off, b_off, w_attn, b_attn, out_offs, attn_ws, loss_acc);
    alpha_kernel<<<dim3(Bb), dim3(TEMB), 0, stream>>>(t, w1, b1, w2, b2, alpha_ws);
    prep_b_kernel<<<dim3(1024), dim3(256), 0, stream>>>(w_fuse, bpack);
    transpose_fc_kernel<<<dim3(Cc/32, HW/32, Bb), dim3(32, 8), 0, stream>>>(Fc, Fct);
    sampler_kernel<<<dim3(Bb*Mm*Hh), dim3(64, 4), 0, stream>>>(
        Fct, out_offs, attn_ws, Wp, head);
    fuse_gemm_kernel<<<dim3(NPIX/256, Cc/64), dim3(256), 0, stream>>>(
        head, bpack, Fx, b_fuse, alpha_ws, out);
    loss_fin_kernel<<<1, 1, 0, stream>>>(loss_acc, out_loss);
}
